// SGReason_24043226923957
// MI455X (gfx1250) — matrix operand order repaired and checked by reading, not compiled, not run on hardware
//
#include <hip/hip_runtime.h>
#include <hip/hip_bf16.h>
#include <math.h>

// ---------------------------------------------------------------------------
// MI455X (gfx1250) SGReason forward pass.
// Heavy GEMMs -> v_wmma_f32_16x16x32_f16, f16 operands staged in LDS,
// f32 accumulate. Weights pre-transposed/converted to f16 once per launch so
// the GEMM hot loop has no bounds checks, no conversions on B, and is
// double-buffered (gload(kt+1) in flight while wmma consumes tile kt).
// ---------------------------------------------------------------------------

typedef _Float16 half_t;
typedef __attribute__((ext_vector_type(16))) _Float16 v16h;
typedef __attribute__((ext_vector_type(8)))  _Float16 v8h;
typedef __attribute__((ext_vector_type(4)))  _Float16 v4h;
typedef __attribute__((ext_vector_type(8)))  float    v8f;

#define LDSTR 40   // LDS row stride in halves: 80B -> every 8-half group 16B aligned

__device__ __forceinline__ float sigm(float x){ return 1.f/(1.f+expf(-x)); }

// Fragment load per CDNA5 WMMA layout: lane = khalf*16 + m16;
// element e holds K = (e/8)*16 + khalf*8 + (e%8)  -> two aligned b128 LDS reads.
__device__ __forceinline__ v16h load_frag(const half_t* rowbase, int khalf){
  const v8h lo = *(const v8h*)(rowbase + khalf*8);
  const v8h hi = *(const v8h*)(rowbase + 16 + khalf*8);
  v16h r;
#pragma unroll
  for(int i=0;i<8;i++){ r[i]=lo[i]; r[8+i]=hi[i]; }
  return r;
}

// ---------------------------------------------------------------------------
// Tiled WMMA GEMM:  C[M,N] = (RS ? diag(rscale)*A : A) @ B + bias
//   A : M x Kp f32 row-major (lda, Kp%32==0, zero padded)
//   BT: N x Kp f16 row-major (pre-transposed weights, ldbt, zero padded)
// Block = 128 threads (4 waves); block tile 64x64; wave tile 32x32 (2x2 WMMA).
// Double-buffered LDS, register-staged global loads.
// ---------------------------------------------------------------------------
template<bool RS>
__global__ __launch_bounds__(128) void k_gemm(
    const float* __restrict__ A, int lda,
    const half_t* __restrict__ BT, int ldbt,
    float* __restrict__ C, int ldc,
    int Kp, const float* __restrict__ bias,
    const float* __restrict__ rscale)
{
  __shared__ __attribute__((aligned(16))) half_t As[2][64*LDSTR];
  __shared__ __attribute__((aligned(16))) half_t Bs[2][64*LDSTR];
  const int tid  = threadIdx.x;
  const int row0 = blockIdx.y*64, col0 = blockIdx.x*64;
  const int lane = tid & 31, wave = tid >> 5;
  const int wr = (wave>>1)*32, wc = (wave&1)*32;
  const int m16 = lane & 15, khalf = lane >> 4;

  // per-thread staging coordinates (fixed across k-tiles)
  int ar_r[4], ar_c[4]; float ar_s[4];
  int br_n[2], br_c[2];
#pragma unroll
  for(int i=0;i<4;i++){
    int e = tid + 128*i;           // 0..511 : A tile 64 rows x 8 float4
    ar_r[i] = e >> 3; ar_c[i] = (e & 7)*4;
    ar_s[i] = RS ? rscale[row0 + ar_r[i]] : 1.f;
  }
#pragma unroll
  for(int i=0;i<2;i++){
    int e = tid + 128*i;           // 0..255 : B tile 64 rows x 4 v8h
    br_n[i] = e >> 2; br_c[i] = (e & 3)*8;
  }

  float4 ar[4]; v8h br[2];
  auto gload = [&](int k0){
#pragma unroll
    for(int i=0;i<4;i++)
      ar[i] = *(const float4*)(A + (size_t)(row0 + ar_r[i])*lda + k0 + ar_c[i]);
#pragma unroll
    for(int i=0;i<2;i++)
      br[i] = *(const v8h*)(BT + (size_t)(col0 + br_n[i])*ldbt + k0 + br_c[i]);
  };
  auto lstore = [&](int buf){
#pragma unroll
    for(int i=0;i<4;i++){
      v4h h; float s = ar_s[i];
      h[0]=(half_t)(ar[i].x*s); h[1]=(half_t)(ar[i].y*s);
      h[2]=(half_t)(ar[i].z*s); h[3]=(half_t)(ar[i].w*s);
      *(v4h*)&As[buf][ar_r[i]*LDSTR + ar_c[i]] = h;
    }
#pragma unroll
    for(int i=0;i<2;i++)
      *(v8h*)&Bs[buf][br_n[i]*LDSTR + br_c[i]] = br[i];
  };

  v8f acc[2][2] = {};
  const int ktiles = Kp >> 5;

  gload(0);
  for(int kt=0; kt<ktiles; kt++){
    const int buf = kt & 1;
    lstore(buf);
    __syncthreads();
    if(kt+1 < ktiles) gload((kt+1) << 5);
    v16h af[2], bf[2];
#pragma unroll
    for(int s=0;s<2;s++) af[s] = load_frag(&As[buf][(wr + s*16 + m16)*LDSTR], khalf);
#pragma unroll
    for(int t=0;t<2;t++) bf[t] = load_frag(&Bs[buf][(wc + t*16 + m16)*LDSTR], khalf);
#pragma unroll
    for(int s=0;s<2;s++)
#pragma unroll
      for(int t=0;t<2;t++)
        acc[s][t] = __builtin_amdgcn_wmma_f32_16x16x32_f16(
            false, af[s], false, bf[t], (short)0, acc[s][t], false, false);
  }

#pragma unroll
  for(int s=0;s<2;s++)
#pragma unroll
    for(int t=0;t<2;t++){
      int col = col0 + wc + t*16 + m16;
      float bv = bias ? bias[col] : 0.f;
#pragma unroll
      for(int r=0;r<8;r++){
        int row = row0 + wr + s*16 + r + 8*khalf;  // D: M = r + 8*khalf, N = m16
        C[(size_t)row*ldc + col] = acc[s][t][r] + bv;
      }
    }
}

// ---------------------------------------------------------------------------
// Relation GEMM with fused neighbor gather + mask + 20/||row|| prescale.
// Row r = [featR[gbase[r]] * scaleF[r] (2048) | lfeats*scaleL (5) | pad (27)]
// Kp = 2080; tiles 0..63 are pure vector gathers, tile 64 handles lfeats+pad.
// ---------------------------------------------------------------------------
__global__ __launch_bounds__(128) void k_gemm_rel(
    const float* __restrict__ featR,
    const int*   __restrict__ gbase,
    const float* __restrict__ scaleF,
    const float* __restrict__ scaleL,
    const float* __restrict__ lf,
    const half_t* __restrict__ BT,   // rel_vis_w^T : 512 x 2080 f16
    float* __restrict__ C,           // 32768 x 512
    const float* __restrict__ bias)
{
  __shared__ __attribute__((aligned(16))) half_t As[2][64*LDSTR];
  __shared__ __attribute__((aligned(16))) half_t Bs[2][64*LDSTR];
  const int Kp = 2080, ldbt = 2080, ldc = 512;
  const int tid  = threadIdx.x;
  const int row0 = blockIdx.y*64, col0 = blockIdx.x*64;
  const int lane = tid & 31, wave = tid >> 5;
  const int wr = (wave>>1)*32, wc = (wave&1)*32;
  const int m16 = lane & 15, khalf = lane >> 4;

  int ar_r[4], ar_c[4]; float ar_s[4], ar_sl[4]; size_t ar_base[4]; int ar_gr[4];
  int br_n[2], br_c[2];
#pragma unroll
  for(int i=0;i<4;i++){
    int e = tid + 128*i;
    ar_r[i] = e >> 3; ar_c[i] = (e & 7)*4;
    int gr = row0 + ar_r[i];
    ar_gr[i]   = gr;
    ar_base[i] = (size_t)gbase[gr];
    ar_s[i]    = scaleF[gr];
    ar_sl[i]   = scaleL[gr];
  }
#pragma unroll
  for(int i=0;i<2;i++){
    int e = tid + 128*i;
    br_n[i] = e >> 2; br_c[i] = (e & 3)*8;
  }

  float4 ar[4]; v8h br[2];
  auto gload = [&](int k0){
    if(k0 < 2048){
#pragma unroll
      for(int i=0;i<4;i++){
        float4 v = *(const float4*)(featR + ar_base[i] + k0 + ar_c[i]);
        float s = ar_s[i];
        v.x*=s; v.y*=s; v.z*=s; v.w*=s;
        ar[i] = v;
      }
    } else {
#pragma unroll
      for(int i=0;i<4;i++){
        float v[4];
#pragma unroll
        for(int j=0;j<4;j++){
          int cc = ar_c[i] + j;     // 0..31 within the tail tile
          v[j] = (cc < 5) ? lf[(size_t)ar_gr[i]*5 + cc] * ar_sl[i] : 0.f;
        }
        ar[i] = make_float4(v[0],v[1],v[2],v[3]);
      }
    }
#pragma unroll
    for(int i=0;i<2;i++)
      br[i] = *(const v8h*)(BT + (size_t)(col0 + br_n[i])*ldbt + k0 + br_c[i]);
  };
  auto lstore = [&](int buf){
#pragma unroll
    for(int i=0;i<4;i++){
      v4h h;
      h[0]=(half_t)ar[i].x; h[1]=(half_t)ar[i].y;
      h[2]=(half_t)ar[i].z; h[3]=(half_t)ar[i].w;
      *(v4h*)&As[buf][ar_r[i]*LDSTR + ar_c[i]] = h;
    }
#pragma unroll
    for(int i=0;i<2;i++)
      *(v8h*)&Bs[buf][br_n[i]*LDSTR + br_c[i]] = br[i];
  };

  v8f acc[2][2] = {};
  const int ktiles = Kp >> 5;  // 65

  gload(0);
  for(int kt=0; kt<ktiles; kt++){
    const int buf = kt & 1;
    lstore(buf);
    __syncthreads();
    if(kt+1 < ktiles) gload((kt+1) << 5);
    v16h af[2], bf[2];
#pragma unroll
    for(int s=0;s<2;s++) af[s] = load_frag(&As[buf][(wr + s*16 + m16)*LDSTR], khalf);
#pragma unroll
    for(int t=0;t<2;t++) bf[t] = load_frag(&Bs[buf][(wc + t*16 + m16)*LDSTR], khalf);
#pragma unroll
    for(int s=0;s<2;s++)
#pragma unroll
      for(int t=0;t<2;t++)
        acc[s][t] = __builtin_amdgcn_wmma_f32_16x16x32_f16(
            false, af[s], false, bf[t], (short)0, acc[s][t], false, false);
  }

#pragma unroll
  for(int s=0;s<2;s++)
#pragma unroll
    for(int t=0;t<2;t++){
      int col = col0 + wc + t*16 + m16;
      float bv = bias[col];
#pragma unroll
      for(int r=0;r<8;r++){
        int row = row0 + wr + s*16 + r + 8*khalf;
        C[(size_t)row*ldc + col] = acc[s][t][r] + bv;
      }
    }
}

// ------------------- weight convert + transpose (once per launch) -----------
// WT[n*Kp + k] = (k<K) ? (f16)W[k*N + n] : 0
__global__ void k_wt(const float* __restrict__ W, half_t* __restrict__ WT,
                     int K, int N, int Kp){
  int i = blockIdx.x*blockDim.x + threadIdx.x;
  if(i >= N*Kp) return;
  int n = i / Kp, k = i - n*Kp;
  WT[i] = (k < K) ? (half_t)W[(size_t)k*N + n] : (half_t)0.f;
}

// --------------------------- small helper kernels ---------------------------

__global__ void k_fill(float* p, int n, float v){
  int i = blockIdx.x*blockDim.x + threadIdx.x;
  if(i < n) p[i] = v;
}

// two dot products per row of h (4096 x 2048): t1 = h.a1, t2 = h.a2
__global__ __launch_bounds__(256) void k_rowdot2(
    const float* __restrict__ h, const float* __restrict__ a1,
    const float* __restrict__ a2, float* t1, float* t2)
{
  int row = blockIdx.x, tid = threadIdx.x;
  const float* hr = h + (size_t)row*2048;
  float s1=0.f, s2=0.f;
  for(int d=tid; d<2048; d+=256){ float v=hr[d]; s1+=v*a1[d]; s2+=v*a2[d]; }
  __shared__ float r1[256], r2[256];
  r1[tid]=s1; r2[tid]=s2; __syncthreads();
  for(int s=128;s;s>>=1){ if(tid<s){ r1[tid]+=r1[tid+s]; r2[tid]+=r2[tid+s]; } __syncthreads(); }
  if(tid==0){ t1[row]=r1[0]; t2[row]=r2[0]; }
}

// GAT: softmax attention over K=8 neighbors + elu aggregation + residual mix
__global__ __launch_bounds__(256) void k_gat_aggr(
    const float* __restrict__ feat, const float* __restrict__ h,
    const float* __restrict__ t1, const float* __restrict__ t2,
    const int* __restrict__ cxt_idx, const int* __restrict__ cxt_mask,
    float* __restrict__ featR)
{
  int row = blockIdx.x, tid = threadIdx.x;
  int b = row >> 6;
  __shared__ float alpha[8];
  __shared__ int   idxs[8];
  if(tid < 8){
    int k = tid;
    int id = cxt_idx[row*8+k];
    int m  = cxt_mask[row*8+k];
    idxs[k] = b*64 + id;
    float e = t1[row] + t2[b*64+id];
    e = (e >= 0.f) ? e : 0.2f*e;            // leaky_relu 0.2
    alpha[k] = (m > 0) ? e : -1e9f;
  }
  __syncthreads();
  if(tid == 0){
    float mx = -1e30f;
    for(int k=0;k<8;k++) mx = fmaxf(mx, alpha[k]);
    float ex[8], sum=0.f;
    for(int k=0;k<8;k++){ ex[k]=expf(alpha[k]-mx); sum+=ex[k]; }
    for(int k=0;k<8;k++) alpha[k]=ex[k]/sum;
  }
  __syncthreads();
  for(int d=tid; d<2048; d+=256){
    float acc = 0.f;
#pragma unroll
    for(int k=0;k<8;k++) acc += alpha[k]*h[(size_t)idxs[k]*2048 + d];
    float nf = (acc > 0.f) ? acc : (expf(acc)-1.f);       // elu
    featR[(size_t)row*2048 + d] = feat[(size_t)row*2048 + d]*0.9f + nf*0.1f;
  }
}

// per-row sum of squares + vis scale 20/(norm+eps)   (cols = 2048)
__global__ __launch_bounds__(256) void k_row_sumsq(
    const float* __restrict__ X, float* sq, float* scale)
{
  int row = blockIdx.x, tid = threadIdx.x;
  const float* x = X + (size_t)row*2048;
  float s = 0.f;
  for(int d=tid; d<2048; d+=256){ float v=x[d]; s+=v*v; }
  __shared__ float red[256];
  red[tid]=s; __syncthreads();
  for(int st=128;st;st>>=1){ if(tid<st) red[tid]+=red[tid+st]; __syncthreads(); }
  if(tid==0){ sq[row]=red[0]; scale[row]=20.f/(sqrtf(red[0])+1e-8f); }
}

// row l2-normalize, cols = 512, in place
__global__ __launch_bounds__(256) void k_norm_rows(float* __restrict__ X){
  int row = blockIdx.x, tid = threadIdx.x;
  float* x = X + (size_t)row*512;
  float a = x[tid], b = x[tid+256];
  __shared__ float red[256];
  red[tid] = a*a + b*b; __syncthreads();
  for(int st=128;st;st>>=1){ if(tid<st) red[tid]+=red[tid+st]; __syncthreads(); }
  float inv = 1.f/(sqrtf(red[0])+1e-8f);
  x[tid] = a*inv; x[tid+256] = b*inv;
}

// word embedding gather, padded to stride 320 (zero pad cols 300..319)
__global__ void k_embed(const int* __restrict__ seq, const float* __restrict__ emb,
                        float* __restrict__ out){
  int i = blockIdx.x*blockDim.x + threadIdx.x;
  if(i >= 8192*320) return;
  int rl = i/320, d = i - rl*320;
  out[i] = (d < 300) ? emb[(size_t)seq[rl]*300 + d] : 0.f;
}

// GRU pointwise update for time step t (rows = 512, H = 512)
__global__ void k_gru_update(const float* __restrict__ gx, const float* __restrict__ gh,
                             float* __restrict__ h, float* __restrict__ ys, int t){
  int i = blockIdx.x*blockDim.x + threadIdx.x;
  if(i >= 512*512) return;
  int row = i >> 9, j = i & 511;
  const float* gxr = gx + (size_t)(row*16 + t)*1536;
  const float* ghr = gh + (size_t)row*1536;
  float r = sigm(gxr[j]        + ghr[j]);
  float z = sigm(gxr[512 + j]  + ghr[512 + j]);
  float c = tanhf(gxr[1024 + j] + r*ghr[1024 + j]);
  float hn = (1.f - z)*c + z*h[i];
  h[i] = hn;
  ys[(size_t)(row*16 + t)*512 + j] = hn;
}

__global__ void k_copy_half(const float* __restrict__ h, float* __restrict__ hidden, int off){
  int i = blockIdx.x*blockDim.x + threadIdx.x;
  if(i >= 512*512) return;
  int row = i >> 9, j = i & 511;
  hidden[(size_t)row*1024 + off + j] = h[i];
}

__global__ void k_wspo(const float* __restrict__ hidden, const float* __restrict__ w,
                       const float* __restrict__ b, float* __restrict__ outw){
  int i = blockIdx.x*blockDim.x + threadIdx.x;
  if(i >= 1536) return;
  int row = i/3, o = i - row*3;
  const float* hr = hidden + (size_t)row*1024;
  float s = b[o];
  for(int d=0; d<1024; d++) s += hr[d]*w[d*3 + o];
  outw[i] = sigm(s);
}

// ModuleInputAttention for node/rel/obj: scores -> masked softmax -> weighted emb
// outputs stored with stride 320 (zero padded) for the following WMMA GEMMs
__global__ __launch_bounds__(128) void k_input_attn(
    const float* __restrict__ of, const float* __restrict__ ob,
    const float* __restrict__ embd, const int* __restrict__ seq,
    const float* __restrict__ wN, const float* __restrict__ wR, const float* __restrict__ wO,
    float* __restrict__ nemb, float* __restrict__ remb, float* __restrict__ oemb)
{
  int row = blockIdx.x, tid = threadIdx.x;
  __shared__ float sc[3][16];
  __shared__ float at[3][16];
  if(tid < 48){
    int x = tid >> 4, l = tid & 15;
    const float* w = (x==0) ? wN : ((x==1) ? wR : wO);
    const float* pf = of + (size_t)(row*16 + l)*512;
    const float* pb = ob + (size_t)(row*16 + l)*512;
    float s = 0.f;
    for(int d=0; d<512; d++) s += pf[d]*w[d] + pb[d]*w[512+d];
    sc[x][l] = s;
  }
  __syncthreads();
  if(tid < 3){
    int x = tid;
    float v[16], mx = -1e30f;
    for(int l=0;l<16;l++){
      float lab = (seq[row*16 + l] != 0) ? 1.f : 0.f;
      float s = (lab > 0.f) ? sc[x][l] : -1e9f;
      v[l] = s; mx = fmaxf(mx, s);
    }
    float sum = 0.f;
    for(int l=0;l<16;l++){ v[l] = expf(v[l]-mx); sum += v[l]; }
    float asum = 0.f;
    for(int l=0;l<16;l++){
      float lab = (seq[row*16 + l] != 0) ? 1.f : 0.f;
      v[l] = v[l]/sum * lab; asum += v[l];
    }
    float inv = 1.f/(asum + 1e-8f);
    for(int l=0;l<16;l++) at[x][l] = v[l]*inv;
  }
  __syncthreads();
  for(int j=tid; j<960; j+=128){
    int x = j/320, d = j - x*320;
    float s = 0.f;
    if(d < 300)
      for(int l=0;l<16;l++) s += at[x][l]*embd[(size_t)(row*16 + l)*320 + d];
    float* outp = (x==0) ? nemb : ((x==1) ? remb : oemb);
    outp[(size_t)row*320 + d] = s;
  }
}

// precompute relation row scales and gather bases
__global__ void k_rel_prep(const int* __restrict__ idx, const int* __restrict__ msk,
                           const float* __restrict__ lf, const float* __restrict__ nodesq,
                           float* __restrict__ sF, float* __restrict__ sL, int* __restrict__ gb){
  int r = blockIdx.x*blockDim.x + threadIdx.x;
  if(r >= 32768) return;
  int b = r >> 9;
  int id = idx[r], m = msk[r];
  float lfsq = 0.f;
#pragma unroll
  for(int j=0;j<5;j++){ float v = lf[(size_t)r*5+j]; lfsq += v*v; }
  float nsq = ((m > 0) ? nodesq[b*64 + id] : 0.f) + lfsq;
  float scv = 20.f/(sqrtf(nsq) + 1e-8f);
  sF[r] = (m > 0) ? scv : 0.f;
  sL[r] = scv;
  gb[r] = (b*64 + id)*2048;
}

// attn_node / attn_obj (one thread per (b,q,n); dot 512 twice)
__global__ void k_attnNO(const float* __restrict__ ve, const float* __restrict__ weN,
                         const float* __restrict__ weO, const int* __restrict__ cls,
                         float* __restrict__ aN, float* __restrict__ aO){
  int i = blockIdx.x*blockDim.x + threadIdx.x;
  if(i >= 32768) return;
  int n = i & 63, q = (i >> 6) & 7, b = i >> 9;
  const float* v  = ve  + (size_t)(b*64 + n)*512;
  const float* w1 = weN + (size_t)(b*8 + q)*512;
  const float* w2 = weO + (size_t)(b*8 + q)*512;
  float s1=0.f, s2=0.f;
  for(int d=0; d<512; d++){ float vv=v[d]; s1+=vv*w1[d]; s2+=vv*w2[d]; }
  float cm = (cls[b*64 + n] != -1) ? 1.f : 0.f;
  aN[i] = s1*cm; aO[i] = s2*cm;
}

// attn_relation (one thread per (b,q,n,k); dot 512)
__global__ void k_attnR(const float* __restrict__ rve, const float* __restrict__ rwe,
                        const int* __restrict__ mask, float* __restrict__ aR){
  int i = blockIdx.x*blockDim.x + threadIdx.x;
  if(i >= 262144) return;
  int nk = i & 511, q = (i >> 9) & 7, b = i >> 12;
  const float* v = rve + (size_t)(b*512 + nk)*512;
  const float* w = rwe + (size_t)(b*8 + q)*512;
  float s = 0.f;
  for(int d=0; d<512; d++) s += v[d]*w[d];
  float m = (mask[b*512 + nk] > 0) ? 1.f : 0.f;
  aR[i] = s*m;
}

// ------------------- sequential transfer/merge loop over NQ -----------------
__device__ __forceinline__ float blockmax64(float v, float* red){
  int n = threadIdx.x;
  red[n] = v; __syncthreads();
  if(n < 32) red[n] = fmaxf(red[n], red[n+32]); __syncthreads();
  if(n < 16) red[n] = fmaxf(red[n], red[n+16]); __syncthreads();
  if(n <  8) red[n] = fmaxf(red[n], red[n+ 8]); __syncthreads();
  if(n <  4) red[n] = fmaxf(red[n], red[n+ 4]); __syncthreads();
  if(n <  2) red[n] = fmaxf(red[n], red[n+ 2]); __syncthreads();
  if(n <  1) red[n] = fmaxf(red[n], red[n+ 1]); __syncthreads();
  float r = red[0]; __syncthreads();
  return r;
}

__global__ __launch_bounds__(64) void k_final(
    const float* __restrict__ attnN, const float* __restrict__ attnO,
    const float* __restrict__ attnR, const float* __restrict__ wspo,
    const int* __restrict__ seq_type, const int* __restrict__ seq_rel,
    const int* __restrict__ com_mask, const int* __restrict__ cxt_idx,
    const int* __restrict__ cxt_mask, const float* __restrict__ gate_node,
    const float* __restrict__ gate_obj, float* __restrict__ out)
{
  int b = blockIdx.x, n = threadIdx.x;
  __shared__ float gsub[8][64], gobj[8][64];
  __shared__ float red[64];
  __shared__ float sobj[64];
  for(int q=0;q<8;q++){ gsub[q][n]=0.f; gobj[q][n]=0.f; }
  float gn0=gate_node[0], gn1=gate_node[1], go0=gate_obj[0], go1=gate_obj[1];
  __syncthreads();

  for(int i=0;i<8;i++){
    float w0 = wspo[(b*8+i)*3 + 0];
    float w2 = wspo[(b*8+i)*3 + 2];
    float an = attnN[(b*8+i)*64 + n];
    float ao = attnO[(b*8+i)*64 + n];

    float s_attn_node = w0 * (an * sigm(gn0*an + gn1));
    float nrm1 = fmaxf(blockmax64(s_attn_node, red), 1.f);
    float s_attn_s = s_attn_node / nrm1;

    float trans = 0.f;
    for(int q=0;q<8;q++){
      int rel = seq_rel[(b*8+i)*8 + q];
      if(rel == 2) trans += gsub[q][n];
      else if(rel == 3) trans += gobj[q][n];
    }
    float gob = ao * sigm(go0*ao + go1) + trans;
    float nrm2 = fmaxf(blockmax64(gob, red), 1.f);
    float sao = gob / nrm2;                 // spo_attn_obj
    sobj[n] = sao;
    __syncthreads();

    float spo_rel = 0.f;
#pragma unroll
    for(int k=0;k<8;k++){
      int idx = cxt_idx[(b*64+n)*8 + k];
      float m = (cxt_mask[(b*64+n)*8 + k] > 0) ? 1.f : 0.f;
      spo_rel += attnR[((size_t)(b*8+i)*64 + n)*8 + k] * sobj[idx] * m;
    }
    float pre = s_attn_node + w2*spo_rel;
    float nrm3 = fmaxf(blockmax64(pre, red), 1.f);
    float spo_attn_s = pre / nrm3;

    int st = seq_type[b*8 + i];
    float ts   = (st == 1) ? 1.f : 0.f;
    float tspo = (st == 0) ? 1.f : 0.f;
    float spo_attn_o = sao * tspo;
    float a_s = s_attn_s*ts + spo_attn_s*tspo;
    float a_o = s_attn_s*ts + spo_attn_o*tspo;   // s_attn_o == s_attn_s

    float mg = 0.f;
    for(int q=0;q<8;q++){
      int rel = seq_rel[(b*8+i)*8 + q];
      if(rel == 0) mg += gsub[q][n];
      else if(rel == 1) mg += gobj[q][n];
    }
    float merged = a_s + mg;
    float nrm4 = fmaxf(blockmax64(merged, red), 1.f);
    a_s = merged / nrm4;

    if(st == 2){ a_s = 0.f; a_o = 0.f; }
    else if(st == -1){ a_s = -1.f; a_o = -1.f; }
    gsub[i][n] = a_s; gobj[i][n] = a_o;
    __syncthreads();
  }

  float sc = 0.f;
  for(int i=0;i<8;i++) sc += (com_mask[b*8+i] == 1) ? gsub[i][n] : 0.f;
  out[b*64 + n] = sc;
}

// ---------------------------------------------------------------------------
extern "C" void kernel_launch(void* const* d_in, const int* in_sizes, int n_in,
                              void* d_out, int out_size, void* d_ws, size_t ws_size,
                              hipStream_t stream)
{
  (void)in_sizes; (void)n_in; (void)out_size; (void)ws_size;
  auto F = [&](int i){ return (const float*)d_in[i]; };
  auto I = [&](int i){ return (const int*)d_in[i]; };

  const float* feature  = F(0);
  const int*   cls      = I(1);
  // d_in[2] lfeat: dead (attn_location deleted)
  const int*   seq      = I(3);
  // d_in[4] seq_weight: unused per reference
  const int*   seq_type = I(5);
  const int*   seq_rel  = I(6);
  const int*   com_mask = I(7);
  const int*   cxt_idx  = I(8);
  const int*   cxt_mask = I(9);
  const float* cxt_lf   = F(10);
  const float* emb      = F(11);
  const float* Wif=F(12), *Whf=F(13), *bif=F(14), *bhf=F(15);
  const float* Wib=F(16), *Whb=F(17), *bib=F(18), *bhb=F(19);
  const float* w_spo=F(20), *b_spo=F(21);
  const float* aw_node=F(22), *aw_rel=F(23), *aw_obj=F(25);
  const float* nvw=F(26), *nvb=F(27), *nww=F(28), *nwb=F(29);
  const float* rvw=F(30), *rvb=F(31), *rww=F(32), *rwb=F(33);
  const float* gat_W=F(38), *gat_a1=F(39), *gat_a2=F(40);
  const float* gate_node=F(41), *gate_obj=F(43);

  char* base = (char*)d_ws; size_t off = 0;
  auto alloc = [&](size_t bytes)->void*{
    void* p = base + off; off = (off + bytes + 255) & ~(size_t)255; return p;
  };
  // f32 buffers
  float* hbuf   = (float*)alloc(4096ull*2048*4);
  float* featR  = (float*)alloc(4096ull*2048*4);
  float* t1     = (float*)alloc(4096*4);
  float* t2     = (float*)alloc(4096*4);
  float* nodesq = (float*)alloc(4096*4);
  float* nscale = (float*)alloc(4096*4);
  float* ve     = (float*)alloc(4096ull*512*4);
  float* rve    = (float*)alloc(32768ull*512*4);
  float* rscF   = (float*)alloc(32768*4);
  float* rscL   = (float*)alloc(32768*4);
  int*   rgb    = (int*)  alloc(32768*4);
  float* embd   = (float*)alloc(8192ull*320*4);   // padded stride 320
  float* gxf    = (float*)alloc(8192ull*1536*4);
  float* gxb    = (float*)alloc(8192ull*1536*4);
  float* gh     = (float*)alloc(512ull*1536*4);
  float* hgru   = (float*)alloc(512ull*512*4);
  float* of     = (float*)alloc(8192ull*512*4);
  float* ob     = (float*)alloc(8192ull*512*4);
  float* hidden = (float*)alloc(512ull*1024*4);
  float* nemb   = (float*)alloc(512ull*320*4);    // padded stride 320
  float* remb   = (float*)alloc(512ull*320*4);
  float* oemb   = (float*)alloc(512ull*320*4);
  float* weN    = (float*)alloc(512ull*512*4);
  float* weO    = (float*)alloc(512ull*512*4);
  float* rwe    = (float*)alloc(512ull*512*4);
  float* wspo   = (float*)alloc(1536*4);
  float* attnN  = (float*)alloc(32768*4);
  float* attnO  = (float*)alloc(32768*4);
  float* attnR  = (float*)alloc(262144*4);
  // f16 transposed weights (N x Kp)
  half_t* gat_WT = (half_t*)alloc(2048ull*2048*2);
  half_t* WifT   = (half_t*)alloc(1536ull*320*2);
  half_t* WibT   = (half_t*)alloc(1536ull*320*2);
  half_t* WhfT   = (half_t*)alloc(1536ull*512*2);
  half_t* WhbT   = (half_t*)alloc(1536ull*512*2);
  half_t* nvwT   = (half_t*)alloc(512ull*2048*2);
  half_t* nwwT   = (half_t*)alloc(512ull*320*2);
  half_t* rvwT   = (half_t*)alloc(512ull*2080*2);
  half_t* rwwT   = (half_t*)alloc(512ull*320*2);

  auto wt = [&](const float* W, half_t* WT, int K, int N, int Kp){
    int tot = N*Kp;
    k_wt<<<(tot+255)/256, 256, 0, stream>>>(W, WT, K, N, Kp);
  };
  auto gemm = [&](const float* A,int lda,const half_t* BT,int ldbt,float* C,int ldc,
                  int M,int N,int Kp,const float* bias,const float* rs){
    dim3 g(N/64, M/64);
    if(rs) k_gemm<true ><<<g, dim3(128), 0, stream>>>(A,lda,BT,ldbt,C,ldc,Kp,bias,rs);
    else   k_gemm<false><<<g, dim3(128), 0, stream>>>(A,lda,BT,ldbt,C,ldc,Kp,bias,nullptr);
  };

  // ---- one-time weight convert/transpose to f16 ----
  wt(gat_W, gat_WT, 2048, 2048, 2048);
  wt(Wif,   WifT,   300,  1536, 320);
  wt(Wib,   WibT,   300,  1536, 320);
  wt(Whf,   WhfT,   512,  1536, 512);
  wt(Whb,   WhbT,   512,  1536, 512);
  wt(nvw,   nvwT,   2048, 512,  2048);
  wt(nww,   nwwT,   300,  512,  320);
  wt(rvw,   rvwT,   2053, 512,  2080);
  wt(rww,   rwwT,   300,  512,  320);

  // ---- GAT ----
  gemm(feature,2048, gat_WT,2048, hbuf,2048, 4096,2048,2048, nullptr,nullptr);
  k_rowdot2<<<4096,256,0,stream>>>(hbuf, gat_a1, gat_a2, t1, t2);
  k_gat_aggr<<<4096,256,0,stream>>>(feature, hbuf, t1, t2, cxt_idx, cxt_mask, featR);
  k_row_sumsq<<<4096,256,0,stream>>>(featR, nodesq, nscale);

  // ---- sequence embedding + input GEMMs ----
  k_embed<<<(8192*320+255)/256,256,0,stream>>>(seq, emb, embd);
  gemm(embd,320, WifT,320, gxf,1536, 8192,1536,320, bif,nullptr);
  gemm(embd,320, WibT,320, gxb,1536, 8192,1536,320, bib,nullptr);

  // ---- GRU forward scan ----
  k_fill<<<1024,256,0,stream>>>(hgru, 512*512, 0.f);
  for(int t=0;t<16;t++){
    gemm(hgru,512, WhfT,512, gh,1536, 512,1536,512, bhf,nullptr);
    k_gru_update<<<1024,256,0,stream>>>(gxf, gh, hgru, of, t);
  }
  k_copy_half<<<1024,256,0,stream>>>(hgru, hidden, 0);

  // ---- GRU backward scan ----
  k_fill<<<1024,256,0,stream>>>(hgru, 512*512, 0.f);
  for(int t=15;t>=0;t--){
    gemm(hgru,512, WhbT,512, gh,1536, 512,1536,512, bhb,nullptr);
    k_gru_update<<<1024,256,0,stream>>>(gxb, gh, hgru, ob, t);
  }
  k_copy_half<<<1024,256,0,stream>>>(hgru, hidden, 512);

  // ---- spo weights + word attentions ----
  k_wspo<<<6,256,0,stream>>>(hidden, w_spo, b_spo, wspo);
  k_input_attn<<<512,128,0,stream>>>(of, ob, embd, seq, aw_node, aw_rel, aw_obj,
                                     nemb, remb, oemb);

  // ---- visual/word embeddings (match3) ----
  gemm(featR,2048, nvwT,2048, ve,512, 4096,512,2048, nvb, nscale);
  k_norm_rows<<<4096,256,0,stream>>>(ve);
  gemm(nemb,320, nwwT,320, weN,512, 512,512,320, nwb,nullptr);
  k_norm_rows<<<512,256,0,stream>>>(weN);
  gemm(oemb,320, nwwT,320, weO,512, 512,512,320, nwb,nullptr);
  k_norm_rows<<<512,256,0,stream>>>(weO);
  gemm(remb,320, rwwT,320, rwe,512, 512,512,320, rwb,nullptr);
  k_norm_rows<<<512,256,0,stream>>>(rwe);

  // ---- relation embedding (fused gather GEMM) ----
  k_rel_prep<<<128,256,0,stream>>>(cxt_idx, cxt_mask, cxt_lf, nodesq, rscF, rscL, rgb);
  {
    dim3 g(512/64, 32768/64);
    k_gemm_rel<<<g, dim3(128), 0, stream>>>(featR, rgb, rscF, rscL, cxt_lf, rvwT, rve, rvb);
  }
  k_norm_rows<<<32768,256,0,stream>>>(rve);

  // ---- attention maps ----
  k_attnNO<<<128,256,0,stream>>>(ve, weN, weO, cls, attnN, attnO);
  k_attnR<<<1024,256,0,stream>>>(rve, rwe, cxt_mask, attnR);

  // ---- sequential transfer/merge + score ----
  k_final<<<64,64,0,stream>>>(attnN, attnO, attnR, wspo, seq_type, seq_rel,
                              com_mask, cxt_idx, cxt_mask, gate_node, gate_obj,
                              (float*)d_out);
}
